// AdvancedFeatureTransformer_44547400794392
// MI455X (gfx1250) — matrix operand order repaired and checked by reading, not compiled, not run on hardware
//
#include <hip/hip_runtime.h>
#include <hip/hip_bf16.h>
#include <math.h>

// ---------------------------------------------------------------------------
// Problem constants (match reference)
// ---------------------------------------------------------------------------
#define BATCH 4096
#define D_IN  512
#define DMODEL 256
#define TOUT  424
#define NLAYER 6
#define LN_EPS 1e-5f

// ---------------------------------------------------------------------------
// Types
// ---------------------------------------------------------------------------
typedef _Float16 v4h  __attribute__((ext_vector_type(4)));
typedef _Float16 v8h  __attribute__((ext_vector_type(8)));
typedef _Float16 v16h __attribute__((ext_vector_type(16)));
typedef float    v8f  __attribute__((ext_vector_type(8)));

// ---------------------------------------------------------------------------
// CDNA5 async memory->LDS staging (GLOBAL_LOAD_ASYNC_TO_LDS_B128, ASYNCcnt)
// ---------------------------------------------------------------------------
__device__ __forceinline__ unsigned lds_off(const void* p) {
    // LDS aperture maps the low 32 bits of the flat address to the LDS offset.
    return (unsigned)(unsigned long long)p;
}
__device__ __forceinline__ void async_copy_b128(unsigned lds_addr, const void* gaddr) {
    asm volatile("global_load_async_to_lds_b128 %0, %1, off"
                 :: "v"(lds_addr), "v"(gaddr) : "memory");
}
// Wait until at most N async-to-LDS ops remain outstanding (in-order returns:
// cnt<=N proves the older double-buffer finished while the newer streams on).
template <int N> __device__ __forceinline__ void wait_async();
template <> __device__ __forceinline__ void wait_async<0>() {
    asm volatile("s_wait_asynccnt 0x0" ::: "memory");
}
template <> __device__ __forceinline__ void wait_async<4>() {
    asm volatile("s_wait_asynccnt 0x4" ::: "memory");
}
template <> __device__ __forceinline__ void wait_async<6>() {
    asm volatile("s_wait_asynccnt 0x6" ::: "memory");
}

// ---------------------------------------------------------------------------
// WMMA fragment helpers (wave32, v_wmma_f32_16x16x32_f16; layouts per ISA 7.12.2)
// ---------------------------------------------------------------------------
// A 16x32: lane holds row M = lane%16; lanes 0-15 K {0-7,16-23}, lanes 16-31
// K {8-15,24-31}.
__device__ __forceinline__ v16h load_frag_a(const _Float16* base, int stride, int lane) {
    const int r  = lane & 15;
    const int kb = (lane < 16) ? 0 : 8;
    const _Float16* p = base + r * stride + kb;
    v8h lo = *(const v8h*)(p);
    v8h hi = *(const v8h*)(p + 16);
    v16h f;
#pragma unroll
    for (int i = 0; i < 8; ++i) { f[i] = lo[i]; f[i + 8] = hi[i]; }
    return f;
}
// B 32x16: lane holds column N = lane%16; lanes 0-15 K {0-7,16-23}, lanes 16-31 K {8-15,24-31}.
// W stored row-major (N x K), so column n of B is row n of W.
__device__ __forceinline__ v16h load_frag_b(const _Float16* base, int stride, int lane) {
    const int n  = lane & 15;
    const int kb = (lane < 16) ? 0 : 8;
    const _Float16* p = base + n * stride + kb;
    v8h lo = *(const v8h*)(p);
    v8h hi = *(const v8h*)(p + 16);
    v16h f;
#pragma unroll
    for (int i = 0; i < 8; ++i) { f[i] = lo[i]; f[i + 8] = hi[i]; }
    return f;
}
__device__ __forceinline__ v8f wmma_f16(v16h a, v16h b, v8f c) {
    return __builtin_amdgcn_wmma_f32_16x16x32_f16(
        false, a, false, b, (short)0, c, false, false);
}
__device__ __forceinline__ float gelu_exact(float x) {
    return 0.5f * x * (1.0f + erff(x * 0.70710678118654752f));
}

// ---------------------------------------------------------------------------
// f32 -> f16 conversion (weights & activations), vectorized 4-wide.
// ---------------------------------------------------------------------------
__global__ __launch_bounds__(256) void cvt_f32_f16_kernel(
    const float* __restrict__ src, _Float16* __restrict__ dst, int n4)
{
    const int i = blockIdx.x * 256 + threadIdx.x;
    if (i < n4) {
        float4 v = ((const float4*)src)[i];
        v4h o;
        o[0] = (_Float16)v.x; o[1] = (_Float16)v.y;
        o[2] = (_Float16)v.z; o[3] = (_Float16)v.w;
        ((v4h*)dst)[i] = o;
    }
}

// ---------------------------------------------------------------------------
// Tiled WMMA GEMM on f16 operands:
//   acc = A(M,K) . W(N,K)^T ; val = act(acc + bias[n]) (+ R[m,n])
//   optionally written to C32 (f32) and/or C16 (f16).
//   64x64 tile / 128 threads (4 waves, each a 32x32 patch, 2x2 frags).
//   Double-buffered async-to-LDS staging: chunk k+1 streams while k computes.
// ---------------------------------------------------------------------------
template <int ACT, bool RES, bool OUT32, bool OUT16>
__global__ __launch_bounds__(128) void gemm_wmma_f16_kernel(
    const _Float16* __restrict__ A, const _Float16* __restrict__ W,
    const float* __restrict__ bias, const float* __restrict__ R,
    float* __restrict__ C32, _Float16* __restrict__ C16,
    int M, int N, int K)
{
    __shared__ __align__(16) _Float16 sA[2][64 * 32];   // 2 x 4KB
    __shared__ __align__(16) _Float16 sB[2][64 * 32];   // 2 x 4KB

    const int tid  = threadIdx.x;
    const int lane = tid & 31;
    const int wave = tid >> 5;
    const int m0   = blockIdx.x * 64;
    const int n0   = blockIdx.y * 64;
    const int wm   = (wave >> 1) * 32;
    const int wn   = (wave & 1) * 32;
    const unsigned ldsA = lds_off(sA);
    const unsigned ldsB = lds_off(sB);

    // Issue one 64x32 f16 tile per operand into buffer `buf` (4 async/wave).
    auto issue_tiles = [&](int buf, int kc) {
#pragma unroll
        for (int i = 0; i < 2; ++i) {
            const int idx = tid + i * 128;
            const int r = idx >> 2, c = idx & 3;
            async_copy_b128(ldsA + buf * 4096 + r * 64 + c * 16,
                            A + (size_t)(m0 + r) * K + kc + c * 8);
            async_copy_b128(ldsB + buf * 4096 + r * 64 + c * 16,
                            W + (size_t)(n0 + r) * K + kc + c * 8);
        }
    };

    v8f acc[2][2] = {};

    issue_tiles(0, 0);
    for (int kc = 0, it = 0; kc < K; kc += 32, ++it) {
        const int cur = it & 1;
        if (kc + 32 < K) { issue_tiles(cur ^ 1, kc + 32); wait_async<4>(); }
        else             { wait_async<0>(); }
        __syncthreads();

        v16h af[2], bf[2];
#pragma unroll
        for (int mt = 0; mt < 2; ++mt)
            af[mt] = load_frag_a(&sA[cur][(wm + mt * 16) * 32], 32, lane);
#pragma unroll
        for (int nt = 0; nt < 2; ++nt)
            bf[nt] = load_frag_b(&sB[cur][(wn + nt * 16) * 32], 32, lane);
#pragma unroll
        for (int mt = 0; mt < 2; ++mt)
#pragma unroll
            for (int nt = 0; nt < 2; ++nt)
                acc[mt][nt] = wmma_f16(af[mt], bf[nt], acc[mt][nt]);
        __syncthreads();
    }

    // Epilogue: C/D layout -> m = vgpr + 8*(lane>=16), n = lane&15.
    const int cn = lane & 15;
    const int mh = (lane >> 4) << 3;
#pragma unroll
    for (int mt = 0; mt < 2; ++mt)
#pragma unroll
        for (int nt = 0; nt < 2; ++nt) {
            const int n = n0 + wn + nt * 16 + cn;
            const float bv = bias[n];
#pragma unroll
            for (int j = 0; j < 8; ++j) {
                const int m = m0 + wm + mt * 16 + mh + j;
                float v = acc[mt][nt][j] + bv;
                if (ACT == 1) v = gelu_exact(v);
                if (RES) v += R[(size_t)m * N + n];
                if (OUT32) C32[(size_t)m * N + n] = v;
                if (OUT16) C16[(size_t)m * N + n] = (_Float16)v;
            }
        }
}

// ---------------------------------------------------------------------------
// LayerNorm over rows of 256 — wave32 per row; writes f32 + f16 copies.
// ---------------------------------------------------------------------------
__global__ __launch_bounds__(128) void ln256_kernel(
    const float* __restrict__ X, const float* __restrict__ g,
    const float* __restrict__ b, float* __restrict__ Y32,
    _Float16* __restrict__ Y16)
{
    const int row  = blockIdx.x * 4 + (threadIdx.x >> 5);
    const int lane = threadIdx.x & 31;
    const float* xr = X + (size_t)row * DMODEL;
    float v[8], s = 0.f, ss = 0.f;
#pragma unroll
    for (int i = 0; i < 8; ++i) {
        v[i] = xr[lane + i * 32];
        s += v[i]; ss += v[i] * v[i];
    }
#pragma unroll
    for (int msk = 16; msk >= 1; msk >>= 1) {
        s  += __shfl_xor(s, msk, 32);
        ss += __shfl_xor(ss, msk, 32);
    }
    const float mean = s * (1.f / DMODEL);
    const float var  = ss * (1.f / DMODEL) - mean * mean;
    const float rstd = rsqrtf(var + LN_EPS);
#pragma unroll
    for (int i = 0; i < 8; ++i) {
        const int c = lane + i * 32;
        const float y = (v[i] - mean) * rstd * g[c] + b[c];
        Y32[(size_t)row * DMODEL + c] = y;
        Y16[(size_t)row * DMODEL + c] = (_Float16)y;
    }
}

// ---------------------------------------------------------------------------
// Fused per-t head: block = (64 batch rows, one t), 128 threads = 4 waves.
//   t1 = h16(64x256) . W1[t](128x256)^T + b1[t]   (WMMA, double-buffered async)
//   t1 = relu(LN_128(t1))                          (in-LDS)
//   t2 = relu(t1(64x128) . W2[t](64x128)^T + b2[t])(WMMA)
//   out[b,t] = t2 . W3[t] + b3[t]
// LDS arena 72KB (well under CDNA5's 320KB/WGP), regions aliased across phases:
//   [0,8K)   sA x2      [8K,24K)  sBc x2      (stage-1 double buffers)
//   [24K,56K) t1 f32    [56K,72K) t1 f16
//   [0,16K)  W2 f16 (after stage 1)   [16K,32K) t2 f32 (after LN)
// ---------------------------------------------------------------------------
__global__ __launch_bounds__(128) void tp_fused_kernel(
    const _Float16* __restrict__ h16,
    const _Float16* __restrict__ W1, const float* __restrict__ b1,
    const float* __restrict__ lng, const float* __restrict__ lnb,
    const _Float16* __restrict__ W2, const float* __restrict__ b2,
    const float* __restrict__ W3, const float* __restrict__ b3,
    float* __restrict__ out)
{
    __shared__ __align__(16) char smem[73728];
    _Float16* sAb  = (_Float16*)(smem);            // 2 x 64x32  (8KB)
    _Float16* sBb  = (_Float16*)(smem + 8192);     // 2 x 128x32 (16KB)
    float*    sT1  = (float*)   (smem + 24576);    // 64x128 f32 (32KB)
    _Float16* sT1h = (_Float16*)(smem + 57344);    // 64x128 f16 (16KB)
    _Float16* sW2  = (_Float16*)(smem);            // 64x128 f16 (reuse)
    float*    sT2  = (float*)   (smem + 16384);    // 64x64  f32 (reuse)

    const int tid  = threadIdx.x;
    const int lane = tid & 31;
    const int wave = tid >> 5;
    const int b0   = blockIdx.x * 64;
    const int t    = blockIdx.y;
    const _Float16* W1t = W1 + (size_t)t * 128 * 256;
    const unsigned ldsA  = lds_off(sAb);
    const unsigned ldsB  = lds_off(sBb);
    const unsigned ldsW2 = lds_off(sW2);

    // Stage-1 chunk staging: 2 A-chunks + 4 B-chunks per thread = 6 async/wave.
    auto issue_s1 = [&](int buf, int kc) {
#pragma unroll
        for (int i = 0; i < 2; ++i) {               // hA 64x32
            const int idx = tid + i * 128;
            const int r = idx >> 2, c = idx & 3;
            async_copy_b128(ldsA + buf * 4096 + r * 64 + c * 16,
                            h16 + (size_t)(b0 + r) * 256 + kc + c * 8);
        }
#pragma unroll
        for (int i = 0; i < 4; ++i) {               // W1 chunk 128x32
            const int idx = tid + i * 128;
            const int r = idx >> 2, c = idx & 3;
            async_copy_b128(ldsB + buf * 8192 + r * 64 + c * 16,
                            W1t + (size_t)r * 256 + kc + c * 8);
        }
    };

    // ---------------- stage 1: t1 = h . W1^T  (M=64, N=128, K=256) ----------
    v8f acc1[4][2] = {};
    const int wn1 = wave * 32;
    issue_s1(0, 0);
    for (int kc = 0, it = 0; kc < 256; kc += 32, ++it) {
        const int cur = it & 1;
        if (kc + 32 < 256) { issue_s1(cur ^ 1, kc + 32); wait_async<6>(); }
        else               { wait_async<0>(); }
        __syncthreads();

        const _Float16* cA = sAb + cur * 2048;
        const _Float16* cB = sBb + cur * 4096;
        v16h bfr[2];
#pragma unroll
        for (int nt = 0; nt < 2; ++nt)
            bfr[nt] = load_frag_b(&cB[(wn1 + nt * 16) * 32], 32, lane);
#pragma unroll
        for (int mt = 0; mt < 4; ++mt) {
            v16h afr = load_frag_a(&cA[(mt * 16) * 32], 32, lane);
#pragma unroll
            for (int nt = 0; nt < 2; ++nt)
                acc1[mt][nt] = wmma_f16(afr, bfr[nt], acc1[mt][nt]);
        }
        __syncthreads();
    }
    // Spill t1 (+bias) to LDS f32.
    {
        const int cn = lane & 15;
        const int mh = (lane >> 4) << 3;
#pragma unroll
        for (int mt = 0; mt < 4; ++mt)
#pragma unroll
            for (int nt = 0; nt < 2; ++nt) {
                const int n = wn1 + nt * 16 + cn;
                const float bb = b1[t * 128 + n];
#pragma unroll
                for (int j = 0; j < 8; ++j)
                    sT1[(mt * 16 + mh + j) * 128 + n] = acc1[mt][nt][j] + bb;
            }
    }
    __syncthreads();

    // ---------------- LayerNorm(128) + ReLU -> f16 ---------------------------
#pragma unroll 1
    for (int rr = 0; rr < 16; ++rr) {
        const int m = wave * 16 + rr;
        float v[4], s = 0.f, ss = 0.f;
#pragma unroll
        for (int i = 0; i < 4; ++i) {
            v[i] = sT1[m * 128 + lane + i * 32];
            s += v[i]; ss += v[i] * v[i];
        }
#pragma unroll
        for (int msk = 16; msk >= 1; msk >>= 1) {
            s  += __shfl_xor(s, msk, 32);
            ss += __shfl_xor(ss, msk, 32);
        }
        const float mean = s * (1.f / 128.f);
        const float var  = ss * (1.f / 128.f) - mean * mean;
        const float rstd = rsqrtf(var + LN_EPS);
#pragma unroll
        for (int i = 0; i < 4; ++i) {
            const int c = lane + i * 32;
            float y = (v[i] - mean) * rstd * lng[t * 128 + c] + lnb[t * 128 + c];
            y = y > 0.f ? y : 0.f;
            sT1h[m * 128 + c] = (_Float16)y;
        }
    }
    __syncthreads();

    // Stage W2[t] 64x128 f16 (16KB) into reused LDS region via async copies.
    {
        const _Float16* W2t = W2 + (size_t)t * 64 * 128;
#pragma unroll
        for (int i = 0; i < 8; ++i) {               // 1024 x 16B chunks
            const int idx = tid + i * 128;
            const int r = idx >> 4, c = idx & 15;
            async_copy_b128(ldsW2 + r * 256 + c * 16,
                            W2t + (size_t)r * 128 + c * 8);
        }
        wait_async<0>();
    }
    __syncthreads();

    // ---------------- stage 2: t2 = relu(t1 . W2^T + b2)  (64x64, K=128) -----
    v8f acc2[2][2] = {};
    const int wm2 = (wave >> 1) * 32;
    const int wn2 = (wave & 1) * 32;
#pragma unroll
    for (int kc = 0; kc < 128; kc += 32) {
        v16h af2[2], bf2[2];
#pragma unroll
        for (int mt = 0; mt < 2; ++mt)
            af2[mt] = load_frag_a(&sT1h[(wm2 + mt * 16) * 128 + kc], 128, lane);
#pragma unroll
        for (int nt = 0; nt < 2; ++nt)
            bf2[nt] = load_frag_b(&sW2[(wn2 + nt * 16) * 128 + kc], 128, lane);
#pragma unroll
        for (int mt = 0; mt < 2; ++mt)
#pragma unroll
            for (int nt = 0; nt < 2; ++nt)
                acc2[mt][nt] = wmma_f16(af2[mt], bf2[nt], acc2[mt][nt]);
    }
    {
        const int cn = lane & 15;
        const int mh = (lane >> 4) << 3;
#pragma unroll
        for (int mt = 0; mt < 2; ++mt)
#pragma unroll
            for (int nt = 0; nt < 2; ++nt) {
                const int n = wn2 + nt * 16 + cn;
                const float bb = b2[t * 64 + n];
#pragma unroll
                for (int j = 0; j < 8; ++j) {
                    float v = acc2[mt][nt][j] + bb;
                    sT2[(wm2 + mt * 16 + mh + j) * 64 + n] = v > 0.f ? v : 0.f;
                }
            }
    }
    __syncthreads();

    // ---------------- stage 3: out[b,t] = t2 . W3[t] + b3[t] -----------------
    if (tid < 64) {
        float a = 0.f;
#pragma unroll
        for (int m2 = 0; m2 < 64; ++m2)
            a += sT2[tid * 64 + m2] * W3[t * 64 + m2];
        out[(size_t)(b0 + tid) * TOUT + t] = a + b3[t];
    }
}

// ---------------------------------------------------------------------------
// Host-side orchestration
// ---------------------------------------------------------------------------
static inline void launch_cvt(const float* s, _Float16* d, size_t n, hipStream_t st) {
    const int n4 = (int)(n / 4);
    cvt_f32_f16_kernel<<<(n4 + 255) / 256, 256, 0, st>>>(s, d, n4);
}

extern "C" void kernel_launch(void* const* d_in, const int* in_sizes, int n_in,
                              void* d_out, int out_size, void* d_ws, size_t ws_size,
                              hipStream_t stream)
{
    const float* x          = (const float*)d_in[0];
    const float* proj_W     = (const float*)d_in[1];
    const float* proj_b     = (const float*)d_in[2];
    const float* attn_in_W  = (const float*)d_in[3];
    const float* attn_in_b  = (const float*)d_in[4];
    const float* attn_out_W = (const float*)d_in[5];
    const float* attn_out_b = (const float*)d_in[6];
    const float* ln_g       = (const float*)d_in[7];
    const float* ln_b       = (const float*)d_in[8];
    const float* ff_W1      = (const float*)d_in[9];
    const float* ff_b1      = (const float*)d_in[10];
    const float* ff_W2      = (const float*)d_in[11];
    const float* ff_b2      = (const float*)d_in[12];
    const float* cross_in_W = (const float*)d_in[13];
    const float* cross_in_b = (const float*)d_in[14];
    const float* cross_out_W= (const float*)d_in[15];
    const float* cross_out_b= (const float*)d_in[16];
    const float* tp_W1      = (const float*)d_in[17];
    const float* tp_b1      = (const float*)d_in[18];
    const float* tp_ln_g    = (const float*)d_in[19];
    const float* tp_ln_b    = (const float*)d_in[20];
    const float* tp_W2      = (const float*)d_in[21];
    const float* tp_b2      = (const float*)d_in[22];
    const float* tp_W3      = (const float*)d_in[23];
    const float* tp_b3      = (const float*)d_in[24];
    float* out = (float*)d_out;

    const int D = DMODEL;

    // ---- workspace carve-out (byte cursor, 16B aligned regions) ----
    char* cur = (char*)d_ws;
    auto take = [&cur](size_t bytes) { char* p = cur; cur += (bytes + 15) & ~(size_t)15; return p; };
    float*    h     = (float*)   take((size_t)BATCH * D * 4);          // 4 MB
    float*    h2    = (float*)   take((size_t)BATCH * D * 4);          // 4 MB
    _Float16* h16   = (_Float16*)take((size_t)BATCH * D * 2);          // 2 MB
    _Float16* tmp16 = (_Float16*)take((size_t)BATCH * 4 * D * 2);      // 8 MB
    _Float16* x16   = (_Float16*)take((size_t)BATCH * D_IN * 2);       // 4 MB
    _Float16* wproj = (_Float16*)take((size_t)D * D_IN * 2);
    _Float16* wain  = (_Float16*)take((size_t)NLAYER * 3 * D * D * 2);
    _Float16* waout = (_Float16*)take((size_t)NLAYER * D * D * 2);
    _Float16* wff1  = (_Float16*)take((size_t)NLAYER * 4 * D * D * 2);
    _Float16* wff2  = (_Float16*)take((size_t)NLAYER * 4 * D * D * 2);
    _Float16* wcin  = (_Float16*)take((size_t)3 * D * D * 2);
    _Float16* wcout = (_Float16*)take((size_t)D * D * 2);
    _Float16* wtp1  = (_Float16*)take((size_t)TOUT * 128 * 256 * 2);   // 27.8 MB
    _Float16* wtp2  = (_Float16*)take((size_t)TOUT * 64 * 128 * 2);    // 6.9 MB

    // ---- one-time (per launch) f32 -> f16 conversions ----
    launch_cvt(x,          x16,   (size_t)BATCH * D_IN, stream);
    launch_cvt(proj_W,     wproj, (size_t)D * D_IN, stream);
    launch_cvt(attn_in_W,  wain,  (size_t)NLAYER * 3 * D * D, stream);
    launch_cvt(attn_out_W, waout, (size_t)NLAYER * D * D, stream);
    launch_cvt(ff_W1,      wff1,  (size_t)NLAYER * 4 * D * D, stream);
    launch_cvt(ff_W2,      wff2,  (size_t)NLAYER * 4 * D * D, stream);
    launch_cvt(cross_in_W, wcin,  (size_t)3 * D * D, stream);
    launch_cvt(cross_out_W,wcout, (size_t)D * D, stream);
    launch_cvt(tp_W1,      wtp1,  (size_t)TOUT * 128 * 256, stream);
    launch_cvt(tp_W2,      wtp2,  (size_t)TOUT * 64 * 128, stream);

    const dim3 blk(128);
    auto grid = [](int M, int N) { return dim3(M / 64, N / 64); };

    // 1) projection: h(+h16) = x16 . proj_W^T + proj_b
    gemm_wmma_f16_kernel<0, false, true, true><<<grid(BATCH, D), blk, 0, stream>>>(
        x16, wproj, proj_b, nullptr, h, h16, BATCH, D, D_IN);

    // 2) backbone layers
    for (int i = 0; i < NLAYER; ++i) {
        const _Float16* Wv = wain + ((size_t)i * 3 * D + 2 * D) * D;
        const float*    bv = attn_in_b + (size_t)i * 3 * D + 2 * D;
        gemm_wmma_f16_kernel<0, false, false, true><<<grid(BATCH, D), blk, 0, stream>>>(
            h16, Wv, bv, nullptr, nullptr, tmp16, BATCH, D, D);
        gemm_wmma_f16_kernel<0, true, true, false><<<grid(BATCH, D), blk, 0, stream>>>(
            tmp16, waout + (size_t)i * D * D, attn_out_b + (size_t)i * D,
            h, h2, nullptr, BATCH, D, D);
        ln256_kernel<<<BATCH / 4, 128, 0, stream>>>(
            h2, ln_g + (size_t)i * D, ln_b + (size_t)i * D, h, h16);
        gemm_wmma_f16_kernel<1, false, false, true><<<grid(BATCH, 4 * D), blk, 0, stream>>>(
            h16, wff1 + (size_t)i * 4 * D * D, ff_b1 + (size_t)i * 4 * D,
            nullptr, nullptr, tmp16, BATCH, 4 * D, D);
        gemm_wmma_f16_kernel<0, true, true, false><<<grid(BATCH, D), blk, 0, stream>>>(
            tmp16, wff2 + (size_t)i * D * 4 * D, ff_b2 + (size_t)i * D,
            h, h2, nullptr, BATCH, D, 4 * D);
        ln256_kernel<<<BATCH / 4, 128, 0, stream>>>(
            h2, ln_g + (size_t)i * D, ln_b + (size_t)i * D, h, h16);
    }

    // 3) cross block: h2 = h + (h . Wv'^T + bv') . Wo^T + bo ; then f16 copy
    gemm_wmma_f16_kernel<0, false, false, true><<<grid(BATCH, D), blk, 0, stream>>>(
        h16, wcin + (size_t)2 * D * D, cross_in_b + 2 * D,
        nullptr, nullptr, tmp16, BATCH, D, D);
    gemm_wmma_f16_kernel<0, true, true, false><<<grid(BATCH, D), blk, 0, stream>>>(
        tmp16, wcout, cross_out_b, h, h2, nullptr, BATCH, D, D);
    launch_cvt(h2, h16, (size_t)BATCH * D, stream);

    // 4) fused per-t head (dominant FLOPs): grid = (64 batch tiles, 424 t's)
    dim3 tpg(BATCH / 64, TOUT);
    tp_fused_kernel<<<tpg, blk, 0, stream>>>(h16, wtp1, tp_b1, tp_ln_g, tp_ln_b,
                                             wtp2, tp_b2, tp_W3, tp_b3, out);
}
